// GraphDepthLSSTransform_45432164057427
// MI455X (gfx1250) — matrix operand order repaired and checked by reading, not compiled, not run on hardware
//
#include <hip/hip_runtime.h>

// ---------------- types ----------------
typedef __attribute__((ext_vector_type(16))) _Float16 v16h;
typedef __attribute__((ext_vector_type(8)))  _Float16 v8h;
typedef __attribute__((ext_vector_type(8)))  float    v8f;
typedef __attribute__((ext_vector_type(4)))  int      v4i;

// CDNA5 async global->LDS path (guarded: falls back to manual staging)
#if defined(__gfx1250__) && __has_builtin(__builtin_amdgcn_global_load_async_to_lds_b128)
#define USE_ASYNC_LDS 1
typedef __attribute__((address_space(1))) v4i* glb_v4ip;   // HIP prints AS(1) as __device__
typedef __attribute__((address_space(3))) v4i* lds_v4ip;   // HIP prints AS(3) as __shared__
#else
#define USE_ASYNC_LDS 0
#endif

__device__ __forceinline__ void wait_asynccnt0() {
#if USE_ASYNC_LDS
#if __has_builtin(__builtin_amdgcn_s_wait_asynccnt)
    __builtin_amdgcn_s_wait_asynccnt(0);
#else
    asm volatile("s_wait_asynccnt 0" ::: "memory");
#endif
#endif
}

// ---------------- problem constants ----------------
#define BN     4
#define C_IN   256
#define C_OUT  80
#define IH     256
#define IW     704
#define FHh    32
#define FWw    88
#define DBINS  118
#define KG     8
#define NXv    360
#define NYv    360
#define NZv    1
#define NPIX   (BN * FHh * FWw)          // 11264
#define PIXB   (FHh * FWw)               // 2816 per image
#define H1     64
#define W1     176
#define NPIX1  (BN * H1 * W1)            // 45056
#define NOUT   (BN * NZv * C_OUT * NXv * NYv)   // 41,472,000
#define N3PAD  256
#define N3VAL  198

// =====================================================================
// zero-fill output
// =====================================================================
__global__ void __launch_bounds__(256) zero_f4(float4* p) {
    size_t i = (size_t)blockIdx.x * 256 + threadIdx.x;
    p[i] = make_float4(0.f, 0.f, 0.f, 0.f);
}

// =====================================================================
// tiny prep kernels
// =====================================================================
__global__ void make_affine(const float* __restrict__ g, const float* __restrict__ be,
                            const float* __restrict__ bconv,
                            float* __restrict__ sc, float* __restrict__ sh, int n) {
    int i = blockIdx.x * 256 + threadIdx.x;
    if (i < n) {
        float s = g[i] * rsqrtf(1.f + 1e-5f);
        sc[i] = s;
        sh[i] = bconv[i] * s + be[i];
    }
}

__global__ void make_bias_affine(const float* __restrict__ b,
                                 float* __restrict__ sc, float* __restrict__ sh,
                                 int npad, int nvalid) {
    int i = blockIdx.x * 256 + threadIdx.x;
    if (i < npad) {
        sc[i] = 1.f;
        sh[i] = (i < nvalid) ? b[i] : 0.f;
    }
}

// OIHW f32 (3x3) -> [tap][in][out] f16
__global__ void __launch_bounds__(256) convert_w3(const float* __restrict__ w,
                                                  _Float16* __restrict__ o,
                                                  int IN, int OUT) {
    int idx = blockIdx.x * 256 + threadIdx.x;
    int total = 9 * IN * OUT;
    if (idx >= total) return;
    int oc  = idx % OUT;
    int i   = (idx / OUT) % IN;
    int tap = idx / (OUT * IN);
    o[idx] = (_Float16)w[((size_t)oc * IN + i) * 9 + tap];
}

// n3 weights (198,256,1,1) -> [256][256] f16, zero-padded columns
__global__ void __launch_bounds__(256) convert_w1(const float* __restrict__ w,
                                                  _Float16* __restrict__ o) {
    int idx = blockIdx.x * 256 + threadIdx.x;   // 256*256
    if (idx >= 256 * N3PAD) return;
    int oc = idx % N3PAD;
    int i  = idx / N3PAD;
    o[idx] = (oc < N3VAL) ? (_Float16)w[(size_t)oc * 256 + i] : (_Float16)0.0f;
}

// combine = rots @ inv(intrins), trans
__global__ void prep_geom(const float* __restrict__ K4, const float* __restrict__ c2l,
                          float* __restrict__ comb, float* __restrict__ tr) {
    int b = threadIdx.x;
    if (b >= BN) return;
    const float* k = K4 + b * 16;
    float a00 = k[0], a01 = k[1], a02 = k[2];
    float a10 = k[4], a11 = k[5], a12 = k[6];
    float a20 = k[8], a21 = k[9], a22 = k[10];
    float det = a00 * (a11 * a22 - a12 * a21) - a01 * (a10 * a22 - a12 * a20)
              + a02 * (a10 * a21 - a11 * a20);
    float id = 1.f / det;
    float inv[9] = {
        (a11 * a22 - a12 * a21) * id, (a02 * a21 - a01 * a22) * id, (a01 * a12 - a02 * a11) * id,
        (a12 * a20 - a10 * a22) * id, (a00 * a22 - a02 * a20) * id, (a02 * a10 - a00 * a12) * id,
        (a10 * a21 - a11 * a20) * id, (a01 * a20 - a00 * a21) * id, (a00 * a11 - a01 * a10) * id };
    const float* m = c2l + b * 16;
    for (int i = 0; i < 3; ++i)
        for (int j = 0; j < 3; ++j)
            comb[b * 9 + i * 3 + j] = m[i * 4 + 0] * inv[0 * 3 + j]
                                    + m[i * 4 + 1] * inv[1 * 3 + j]
                                    + m[i * 4 + 2] * inv[2 * 3 + j];
    for (int i = 0; i < 3; ++i) tr[b * 3 + i] = m[i * 4 + 3];
}

// =====================================================================
// d1: fused c1+c2 (1x1) + bn_relu + 5x5 s4 p2 conv (16->32) + bn_relu
// =====================================================================
__global__ void __launch_bounds__(256) d1_fused(
    const float* __restrict__ depth, const float* __restrict__ nbd,
    const float* __restrict__ c1w, const float* __restrict__ c1s, const float* __restrict__ c1sh,
    const float* __restrict__ c2w, const float* __restrict__ c2s, const float* __restrict__ c2sh,
    const float* __restrict__ d1w, const float* __restrict__ d1s, const float* __restrict__ d1sh,
    float* __restrict__ out /* [NPIX1][32] */) {
    __shared__ float wl[25 * 16 * 32];           // 51.2 KB
    for (int i = threadIdx.x; i < 25 * 16 * 32; i += 256) {
        int oc = i & 31, ic = (i >> 5) & 15, t = i >> 9;
        wl[i] = d1w[((size_t)oc * 16 + ic) * 25 + t];
    }
    __syncthreads();

    int g = blockIdx.x * 256 + threadIdx.x;      // 45056 total
    int b = g / (H1 * W1);
    int r = g % (H1 * W1);
    int oy = r / W1, ox = r % W1;

    float acc[32];
#pragma unroll
    for (int i = 0; i < 32; ++i) acc[i] = 0.f;

    for (int ky = 0; ky < 5; ++ky) {
        int iy = oy * 4 + ky - 2;
        if (iy < 0 || iy >= IH) continue;
        for (int kx = 0; kx < 5; ++kx) {
            int ix = ox * 4 + kx - 2;
            if (ix < 0 || ix >= IW) continue;
            int t = ky * 5 + kx;
            float v[16];
            float dv = depth[((size_t)b * IH + iy) * IW + ix];
#pragma unroll
            for (int i = 0; i < 8; ++i)
                v[i] = fmaxf(c1s[i] * (c1w[i] * dv) + c1sh[i], 0.f);
#pragma unroll
            for (int i = 0; i < 8; ++i) {
                float s = 0.f;
#pragma unroll
                for (int j = 0; j < KG; ++j)
                    s += c2w[i * KG + j] * nbd[(((size_t)b * KG + j) * IH + iy) * IW + ix];
                v[8 + i] = fmaxf(c2s[i] * s + c2sh[i], 0.f);
            }
            const float* wt = &wl[t * 16 * 32];
#pragma unroll
            for (int ic = 0; ic < 16; ++ic) {
                float vv = v[ic];
#pragma unroll
                for (int oc = 0; oc < 32; ++oc) acc[oc] += wt[ic * 32 + oc] * vv;
            }
        }
    }
#pragma unroll
    for (int oc = 0; oc < 32; ++oc)
        out[(size_t)g * 32 + oc] = fmaxf(d1s[oc] * acc[oc] + d1sh[oc], 0.f);
}

// =====================================================================
// d2: 5x5 s2 p2 conv 32->64 + bn_relu -> hbuf channels 0..63 (f16)
// =====================================================================
__global__ void __launch_bounds__(256) d2_conv(
    const float* __restrict__ in /* [NPIX1][32] */,
    const float* __restrict__ d2w, const float* __restrict__ d2s, const float* __restrict__ d2sh,
    _Float16* __restrict__ hbuf /* [NPIX][320] */) {
    int oc0 = blockIdx.y * 16;
    __shared__ float wl[25 * 32 * 16];           // 51.2 KB
    for (int i = threadIdx.x; i < 25 * 32 * 16; i += 256) {
        int oc = i & 15, ic = (i >> 4) & 31, t = i >> 9;
        wl[i] = d2w[(((size_t)(oc0 + oc) * 32) + ic) * 25 + t];
    }
    __syncthreads();

    int g = blockIdx.x * 256 + threadIdx.x;      // 11264 total
    int b = g / PIXB;
    int r = g % PIXB;
    int oy = r / FWw, ox = r % FWw;

    float acc[16];
#pragma unroll
    for (int i = 0; i < 16; ++i) acc[i] = 0.f;

    for (int ky = 0; ky < 5; ++ky) {
        int iy = oy * 2 + ky - 2;
        if (iy < 0 || iy >= H1) continue;
        for (int kx = 0; kx < 5; ++kx) {
            int ix = ox * 2 + kx - 2;
            if (ix < 0 || ix >= W1) continue;
            int t = ky * 5 + kx;
            const float* src = &in[(((size_t)b * H1 + iy) * W1 + ix) * 32];
            const float* wt = &wl[t * 32 * 16];
#pragma unroll
            for (int ic = 0; ic < 32; ++ic) {
                float vv = src[ic];
#pragma unroll
                for (int oc = 0; oc < 16; ++oc) acc[oc] += wt[ic * 16 + oc] * vv;
            }
        }
    }
#pragma unroll
    for (int oc = 0; oc < 16; ++oc) {
        float v = fmaxf(d2s[oc0 + oc] * acc[oc] + d2sh[oc0 + oc], 0.f);
        hbuf[(size_t)g * 320 + oc0 + oc] = (_Float16)v;
    }
}

// img (B,256,32,88) f32 NCHW -> hbuf channels 64..319 f16
__global__ void __launch_bounds__(256) img_to_h(const float* __restrict__ img,
                                                _Float16* __restrict__ hbuf) {
    size_t idx = (size_t)blockIdx.x * 256 + threadIdx.x;   // BN*256*2816
    size_t b = idx / ((size_t)C_IN * PIXB);
    size_t r = idx % ((size_t)C_IN * PIXB);
    size_t c = r / PIXB;
    size_t p = r % PIXB;
    hbuf[(b * PIXB + p) * 320 + 64 + c] = (_Float16)img[idx];
}

// =====================================================================
// WMMA implicit-GEMM conv, double-buffered LDS, 1 barrier per K-step.
// Block = 256 threads = 8 wave32s. Block tile: 256 pixels x 64 out ch.
// Each wave: 32 pixels x 64 channels = 2M x 4N accumulators
//            -> 8x v_wmma_f32_16x16x32_f16 per K-step.
// A tile staged with GLOBAL_LOAD_ASYNC_TO_LDS_B128 when available.
// in: [NPIX][IN_CH] f16 NHWC, w: [KS*KS][IN_CH][OUT_PAD] f16
// =====================================================================
#define ASTR 40
#define BSTR 40

__device__ __forceinline__ v16h ld_frag16(const _Float16* p0, const _Float16* p1) {
    v8h a = *(const v8h*)p0;
    v8h b = *(const v8h*)p1;
    return __builtin_shufflevector(a, b, 0, 1, 2, 3, 4, 5, 6, 7,
                                   8, 9, 10, 11, 12, 13, 14, 15);
}

template <int IN_CH, int OUT_PAD, int KS, bool RELU, bool OUT_F16>
__global__ void __launch_bounds__(256) conv_wmma(
    const _Float16* __restrict__ in, const _Float16* __restrict__ w,
    const float* __restrict__ scale, const float* __restrict__ shift,
    void* __restrict__ outv, int out_stride, int out_valid) {
    constexpr int KCH = IN_CH / 32;        // K chunks per tap
    constexpr int NIT = KS * KS * KCH;     // total K-steps

    __shared__ __align__(16) _Float16 Alds[2][256 * ASTR];  // 2 x 20 KB
    __shared__ __align__(16) _Float16 Blds[2][64 * BSTR];   // 2 x 5 KB

    const int tid  = threadIdx.x;
    const int lane = tid & 31;
    const int wid  = tid >> 5;            // 8 waves -> M offset wid*32
    const int hf   = lane >> 4;
    const int l16  = lane & 15;
    const int Mbase = blockIdx.x * 256;
    const int Nblk  = blockIdx.y * 64;

    // A staging: one thread per pixel row, 32 contiguous f16 channels
    const int mg = Mbase + tid;
    const int rr = mg % PIXB;
    const int py = rr / FWw;
    const int px = rr % FWw;
    // B staging: k = tid>>3 (0..31), 8 columns per thread
    const int bk  = tid >> 3;
    const int bn0 = (tid & 7) * 8;

    v8f acc[8];                           // [tm*4 + nn]
#pragma unroll
    for (int i = 0; i < 8; ++i)
#pragma unroll
        for (int j = 0; j < 8; ++j) acc[i][j] = 0.f;

    auto stage = [&](int it, int buf) {
        const int tap = it / KCH;
        const int kc  = (it - tap * KCH) * 32;
        const int dy  = tap / KS - (KS / 2);
        const int dx  = tap % KS - (KS / 2);
        const int yy  = py + dy;
        const int xx  = px + dx;
        const bool inb = (yy >= 0) && (yy < FHh) && (xx >= 0) && (xx < FWw);
        const _Float16* asrc = in + ((size_t)(mg + dy * FWw + dx)) * IN_CH + kc;
#if USE_ASYNC_LDS
        if (inb) {
#pragma unroll
            for (int q = 0; q < 4; ++q) {
                __builtin_amdgcn_global_load_async_to_lds_b128(
                    (glb_v4ip)(void*)(asrc + q * 8),
                    (lds_v4ip)(&Alds[buf][tid * ASTR + q * 8]),
                    0, 0);
            }
        } else {
#pragma unroll
            for (int q = 0; q < 4; ++q) {
                v8h z;
#pragma unroll
                for (int i = 0; i < 8; ++i) z[i] = (_Float16)0.0f;
                *(v8h*)(&Alds[buf][tid * ASTR + q * 8]) = z;
            }
        }
#else
#pragma unroll
        for (int q = 0; q < 4; ++q) {
            v8h v;
            if (inb) {
                v = *(const v8h*)(asrc + q * 8);
            } else {
#pragma unroll
                for (int i = 0; i < 8; ++i) v[i] = (_Float16)0.0f;
            }
            *(v8h*)(&Alds[buf][tid * ASTR + q * 8]) = v;
        }
#endif
        const _Float16* ws = w + ((size_t)(tap * IN_CH + kc + bk)) * OUT_PAD + Nblk + bn0;
        v8h bw = *(const v8h*)ws;
#pragma unroll
        for (int i = 0; i < 8; ++i) Blds[buf][(bn0 + i) * BSTR + bk] = bw[i];
    };

    stage(0, 0);
    for (int it = 0; it < NIT; ++it) {
        wait_asynccnt0();                 // tile staged last iter is in LDS
        __syncthreads();
        if (it + 1 < NIT) stage(it + 1, (it + 1) & 1);
        const int buf = it & 1;

        // A fragments: rows wid*32 + {0,16} + l16 ; K = (j/8)*16 + hf*8 + j%8
        const _Float16* ar0 = &Alds[buf][(wid * 32 + l16) * ASTR];
        const _Float16* ar1 = &Alds[buf][(wid * 32 + 16 + l16) * ASTR];
        v16h A0 = ld_frag16(ar0 + hf * 8, ar0 + 16 + hf * 8);
        v16h A1 = ld_frag16(ar1 + hf * 8, ar1 + 16 + hf * 8);
#pragma unroll
        for (int nn = 0; nn < 4; ++nn) {
            // B fragment: col n = nn*16 + l16 ; K = (j/8)*16 + hf*8 + j%8 (transposed LDS)
            const _Float16* br = &Blds[buf][(nn * 16 + l16) * BSTR + hf * 8];
            v16h Bf = ld_frag16(br, br + 16);
            acc[nn] = __builtin_amdgcn_wmma_f32_16x16x32_f16(
                false, A0, false, Bf, (short)0, acc[nn], false, false);
            acc[4 + nn] = __builtin_amdgcn_wmma_f32_16x16x32_f16(
                false, A1, false, Bf, (short)0, acc[4 + nn], false, false);
        }
    }

    // ---- epilogue: fused BN + ReLU ----
#pragma unroll
    for (int nn = 0; nn < 4; ++nn) {
        const int n = Nblk + nn * 16 + l16;
        const float sc = scale[n];
        const float sh = shift[n];
        const bool nok = (n < out_valid);
#pragma unroll
        for (int tm = 0; tm < 2; ++tm) {
#pragma unroll
            for (int r = 0; r < 8; ++r) {
                int m = Mbase + wid * 32 + tm * 16 + hf * 8 + r;
                float v = acc[tm * 4 + nn][r] * sc + sh;
                if (RELU) v = fmaxf(v, 0.f);
                if (nok) {
                    if (OUT_F16)
                        ((_Float16*)outv)[(size_t)m * out_stride + n] = (_Float16)v;
                    else
                        ((float*)outv)[(size_t)m * out_stride + n] = v;
                }
            }
        }
    }
}

// =====================================================================
// softmax over 118 depth bins + frustum geometry + voxel scatter
// =====================================================================
__global__ void __launch_bounds__(128) softmax_scatter(
    const float* __restrict__ logits /* [NPIX][198] */,
    const float* __restrict__ comb, const float* __restrict__ tr,
    float* __restrict__ out) {
    __shared__ float sd[DBINS];
    __shared__ float sctx[C_OUT];
    __shared__ int   sbase[DBINS];
    __shared__ float sinv;

    const int g = blockIdx.x;
    const int t = threadIdx.x;
    const int b = g / PIXB;
    const int r = g % PIXB;
    const int y = r / FWw;
    const int x = r % FWw;

    if (t < DBINS) sd[t] = logits[(size_t)g * N3VAL + t];
    if (t < C_OUT) sctx[t] = logits[(size_t)g * N3VAL + DBINS + t];
    __syncthreads();

    if (t == 0) {
        float mx = -3.4e38f;
        for (int i = 0; i < DBINS; ++i) mx = fmaxf(mx, sd[i]);
        float s = 0.f;
        for (int i = 0; i < DBINS; ++i) {
            float e = __expf(sd[i] - mx);
            sd[i] = e;
            s += e;
        }
        sinv = 1.f / s;
    }
    if (t < DBINS) {
        const float* cb = comb + b * 9;
        float Z  = 1.0f + 0.5f * (float)t;
        float Xi = (float)x * (703.0f / 87.0f);
        float Yi = (float)y * (255.0f / 31.0f);
        float p0 = Xi * Z, p1 = Yi * Z, p2 = Z;
        float g0 = cb[0] * p0 + cb[1] * p1 + cb[2] * p2 + tr[b * 3 + 0];
        float g1 = cb[3] * p0 + cb[4] * p1 + cb[5] * p2 + tr[b * 3 + 1];
        float g2 = cb[6] * p0 + cb[7] * p1 + cb[8] * p2 + tr[b * 3 + 2];
        int cx = (int)((g0 + 54.0f) * (1.0f / 0.3f));
        int cy = (int)((g1 + 54.0f) * (1.0f / 0.3f));
        int cz = (int)((g2 + 10.0f) * (1.0f / 20.0f));
        bool kept = (g0 + 54.0f >= 0.f) && (cx >= 0) && (cx < NXv) &&
                    (g1 + 54.0f >= 0.f) && (cy >= 0) && (cy < NYv) &&
                    (g2 + 10.0f >= 0.f) && (cz >= 0) && (cz < NZv);
        sbase[t] = kept
            ? ((b * NZv + cz) * C_OUT) * (NXv * NYv) + cx * NYv + cy
            : -1;
    }
    __syncthreads();

    const float inv = sinv;
    for (int i = t; i < DBINS * C_OUT; i += 128) {
        int d = i / C_OUT;
        int c = i - d * C_OUT;
        int bs = sbase[d];
        if (bs >= 0)
            atomicAdd(&out[(size_t)bs + (size_t)c * (NXv * NYv)],
                      sd[d] * inv * sctx[c]);
    }
}

// =====================================================================
// host launcher
// =====================================================================
static inline char* carve(char*& p, size_t bytes) {
    char* r = p;
    p += (bytes + 255) & ~(size_t)255;
    return r;
}

extern "C" void kernel_launch(void* const* d_in, const int* in_sizes, int n_in,
                              void* d_out, int out_size, void* d_ws, size_t ws_size,
                              hipStream_t stream) {
    (void)in_sizes; (void)n_in; (void)out_size; (void)ws_size;

    const float* img   = (const float*)d_in[0];
    const float* depth = (const float*)d_in[1];
    const float* nbd   = (const float*)d_in[2];
    const float* camK  = (const float*)d_in[3];
    const float* c2l   = (const float*)d_in[4];
    const float* c1w = (const float*)d_in[5],  *c1b = (const float*)d_in[6];
    const float* c1g = (const float*)d_in[7],  *c1be = (const float*)d_in[8];
    const float* c2w = (const float*)d_in[9],  *c2b = (const float*)d_in[10];
    const float* c2g = (const float*)d_in[11], *c2be = (const float*)d_in[12];
    const float* d1w = (const float*)d_in[13], *d1b = (const float*)d_in[14];
    const float* d1g = (const float*)d_in[15], *d1be = (const float*)d_in[16];
    const float* d2w = (const float*)d_in[17], *d2b = (const float*)d_in[18];
    const float* d2g = (const float*)d_in[19], *d2be = (const float*)d_in[20];
    const float* n1w = (const float*)d_in[21], *n1b = (const float*)d_in[22];
    const float* n1g = (const float*)d_in[23], *n1be = (const float*)d_in[24];
    const float* n2w = (const float*)d_in[25], *n2b = (const float*)d_in[26];
    const float* n2g = (const float*)d_in[27], *n2be = (const float*)d_in[28];
    const float* n3w = (const float*)d_in[29], *n3b = (const float*)d_in[30];

    char* p = (char*)d_ws;
    float*    d1out  = (float*)   carve(p, (size_t)NPIX1 * 32 * 4);
    _Float16* hbuf   = (_Float16*)carve(p, (size_t)NPIX * 320 * 2);
    _Float16* a1     = (_Float16*)carve(p, (size_t)NPIX * 256 * 2);
    _Float16* a2     = (_Float16*)carve(p, (size_t)NPIX * 256 * 2);
    float*    logits = (float*)   carve(p, (size_t)NPIX * N3VAL * 4);
    _Float16* w1     = (_Float16*)carve(p, (size_t)9 * 320 * 256 * 2);
    _Float16* w2     = (_Float16*)carve(p, (size_t)9 * 256 * 256 * 2);
    _Float16* w3     = (_Float16*)carve(p, (size_t)256 * N3PAD * 2);
    float* c1s  = (float*)carve(p, 8 * 4);    float* c1sh = (float*)carve(p, 8 * 4);
    float* c2s  = (float*)carve(p, 8 * 4);    float* c2sh = (float*)carve(p, 8 * 4);
    float* d1s  = (float*)carve(p, 32 * 4);   float* d1sh = (float*)carve(p, 32 * 4);
    float* d2s  = (float*)carve(p, 64 * 4);   float* d2sh = (float*)carve(p, 64 * 4);
    float* n1s  = (float*)carve(p, 256 * 4);  float* n1sh = (float*)carve(p, 256 * 4);
    float* n2s  = (float*)carve(p, 256 * 4);  float* n2sh = (float*)carve(p, 256 * 4);
    float* n3s  = (float*)carve(p, N3PAD * 4);float* n3sh = (float*)carve(p, N3PAD * 4);
    float* comb = (float*)carve(p, BN * 9 * 4);
    float* trn  = (float*)carve(p, BN * 3 * 4);

    // 0) zero output (166 MB)
    zero_f4<<<NOUT / 4 / 256, 256, 0, stream>>>((float4*)d_out);

    // 1) parameter prep
    make_affine<<<1, 256, 0, stream>>>(c1g, c1be, c1b, c1s, c1sh, 8);
    make_affine<<<1, 256, 0, stream>>>(c2g, c2be, c2b, c2s, c2sh, 8);
    make_affine<<<1, 256, 0, stream>>>(d1g, d1be, d1b, d1s, d1sh, 32);
    make_affine<<<1, 256, 0, stream>>>(d2g, d2be, d2b, d2s, d2sh, 64);
    make_affine<<<1, 256, 0, stream>>>(n1g, n1be, n1b, n1s, n1sh, 256);
    make_affine<<<1, 256, 0, stream>>>(n2g, n2be, n2b, n2s, n2sh, 256);
    make_bias_affine<<<1, 256, 0, stream>>>(n3b, n3s, n3sh, N3PAD, N3VAL);
    convert_w3<<<(9 * 320 * 256 + 255) / 256, 256, 0, stream>>>(n1w, w1, 320, 256);
    convert_w3<<<(9 * 256 * 256 + 255) / 256, 256, 0, stream>>>(n2w, w2, 256, 256);
    convert_w1<<<(256 * N3PAD + 255) / 256, 256, 0, stream>>>(n3w, w3);
    prep_geom<<<1, BN, 0, stream>>>(camK, c2l, comb, trn);

    // 2) depth path (fused c1/c2 -> d1 -> d2 -> hbuf[0:64]), img -> hbuf[64:320]
    d1_fused<<<NPIX1 / 256, 256, 0, stream>>>(depth, nbd, c1w, c1s, c1sh,
                                              c2w, c2s, c2sh, d1w, d1s, d1sh, d1out);
    d2_conv<<<dim3(NPIX / 256, 4), 256, 0, stream>>>(d1out, d2w, d2s, d2sh, hbuf);
    img_to_h<<<(BN * C_IN * PIXB) / 256, 256, 0, stream>>>(img, hbuf);

    // 3) WMMA implicit-GEMM convs (block tile 256 px x 64 ch, dbl-buffered)
    conv_wmma<320, 256, 3, true, true><<<dim3(NPIX / 256, 256 / 64), 256, 0, stream>>>(
        hbuf, w1, n1s, n1sh, (void*)a1, 256, 256);
    conv_wmma<256, 256, 3, true, true><<<dim3(NPIX / 256, 256 / 64), 256, 0, stream>>>(
        a1, w2, n2s, n2sh, (void*)a2, 256, 256);
    conv_wmma<256, N3PAD, 1, false, false><<<dim3(NPIX / 256, N3PAD / 64), 256, 0, stream>>>(
        a2, w3, n3s, n3sh, (void*)logits, N3VAL, N3VAL);

    // 4) softmax + frustum geometry + voxel-pool scatter
    softmax_scatter<<<NPIX, 128, 0, stream>>>(logits, comb, trn, (float*)d_out);
}
